// Seq2Seq_TeacherForcing_71055938945766
// MI455X (gfx1250) — matrix operand order repaired and checked by reading, not compiled, not run on hardware
//
#include <hip/hip_runtime.h>
#include <hip/hip_bf16.h>

// Seq2Seq LSTM (B=128, T=256, D=64, H=1024) on gfx1250.
// bf16 WMMA for every GEMM, fp32 cell state, bf16 weights staged per-block
// into LDS by the Tensor Data Mover (double-buffered K chunks, TENSORcnt
// overlap). One kernel launch per LSTM cell per timestep; stream order gives
// inter-layer/inter-step sync; h is double-buffered across steps.

#define DEV_INLINE __device__ __forceinline__

typedef __attribute__((ext_vector_type(16))) __bf16 v16bf;
typedef __attribute__((ext_vector_type(8)))  __bf16 v8bf;
typedef __attribute__((ext_vector_type(8)))  float  v8f;
typedef __attribute__((ext_vector_type(4)))  unsigned int v4u;
typedef __attribute__((ext_vector_type(4)))  int v4i;
typedef __attribute__((ext_vector_type(8)))  int v8i;

#define B_ 128
#define T_ 256
#define D_ 64
#define H_ 1024

// K-chunk staged per TDM op: 4 gates x 16 cols x KCHUNK k x 2B (+16B row pad)
#define KCHUNK 128
#define SMEM_BYTES (2 * 64 * (2 * KCHUNK + 16))   // two buffers, 17KB each

#if defined(__AMDGCN__) && __has_builtin(__builtin_amdgcn_tensor_load_to_lds) && \
    __has_builtin(__builtin_amdgcn_s_wait_tensorcnt)
#define HAVE_TDM 1
#else
#define HAVE_TDM 0
#endif

DEV_INLINE __bf16 f2bf(float f) {
  unsigned u = __float_as_uint(f);
  unsigned r = (u + 0x7FFFu + ((u >> 16) & 1u)) >> 16;
  unsigned short s = (unsigned short)r;
  return __builtin_bit_cast(__bf16, s);
}

DEV_INLINE float sigm(float x) { return 1.0f / (1.0f + __expf(-x)); }

// A fragment: 16x32 (MxK) bf16. Lanes 0-15: rows M, K{0..15};
// lanes 16-31: K{16..31}. Two contiguous 16B loads per lane.
DEV_INLINE v16bf load_a_frag(const __bf16* __restrict__ base, int lda,
                             int row, int k0, int lane) {
  const int m    = row + (lane & 15);
  const int off8 = (lane >> 4) << 4;
  const __bf16* p = base + (size_t)m * lda + k0 + off8;
  v8bf lo = *(const v8bf*)(p);
  v8bf hi = *(const v8bf*)(p + 8);
  v16bf r;
#pragma unroll
  for (int i = 0; i < 8; ++i) { r[i] = lo[i]; r[i + 8] = hi[i]; }
  return r;
}

#if HAVE_TDM
// One 3D TDM descriptor loads a 4(gate) x 16(col) x kchunk(K) bf16 tile:
//   addr = ga + 2*(x + y*ldw + z*(H*ldw)),  tile = (kchunk, 16, 4)
// LDS pad: +4 DWORDs after every row (row = kchunk elems) -> rowStride
// = kchunk*2+16 bytes, keeps 16B alignment and staggers banks lane-to-lane.
DEV_INLINE void tdm_load_tile(const __bf16* gsrc, unsigned ldsAddr,
                              int kchunk, int ldw) {
  const unsigned long long ga = (unsigned long long)gsrc;
  v4u g0;
  g0[0] = 1u;                                   // count=1 (valid), user mode
  g0[1] = ldsAddr;                              // LDS byte address of dest
  g0[2] = (unsigned)ga;                         // global_addr[31:0]
  g0[3] = (unsigned)((ga >> 32) & 0x01ffffffull) | (2u << 30);  // [56:32]|type=2

  const unsigned padInt = (kchunk == 128) ? 5u : 4u;  // 64 / 32 dwords per row
  const unsigned td0 = (unsigned)kchunk;
  const unsigned long long s0 = (unsigned long long)ldw;        // y: row stride
  const unsigned long long s1 = (unsigned long long)ldw * H_;   // z: gate stride
  v8i g1;
  g1[0] = (int)((1u << 16)            // data_size = 2 bytes
              | (1u << 20)            // pad_enable
              | (padInt << 22)        // pad_interval = one row
              | (3u << 25));          // pad_amount = 4 dwords
  g1[1] = (int)((td0 & 0xffffu) << 16);                       // tensor_dim0 lo
  g1[2] = (int)(((td0 >> 16) & 0xffffu) | (16u << 16));       // tensor_dim1=16
  g1[3] = (int)(td0 << 16);                                   // tile_dim0
  g1[4] = (int)(16u | (4u << 16));                            // tile1=16 tile2=4
  g1[5] = (int)(s0 & 0xffffffffull);
  g1[6] = (int)(((s0 >> 32) & 0xffffull) | ((s1 & 0xffffull) << 16));
  g1[7] = (int)((s1 >> 16) & 0xffffffffull);

  v4i g2 = {4, 1, 0, 0};   // tensor_dim2=4, tensor_dim3=1, dim2_stride=0
  v4i g3 = {0, 0, 0, 0};

#if defined(__clang_major__) && __clang_major__ >= 23
  v8i g4 = {0, 0, 0, 0, 0, 0, 0, 0};
  __builtin_amdgcn_tensor_load_to_lds(g0, g1, g2, g3, g4, 0);
#else
  __builtin_amdgcn_tensor_load_to_lds(g0, g1, g2, g3, 0);
#endif
}
#endif  // HAVE_TDM

// WMMA over one staged K chunk. B fragment from LDS: lane holds column
// r=lane&15, K block koff = 0/16 elems -> two 16B ds loads per gate.
DEV_INLINE void compute_chunk(const __bf16* __restrict__ A, int lda, int kbase,
                              int kchunk, const char* buf, int rowStride,
                              int rowb, int lane, v8f acc[4]) {
  const int r    = lane & 15;
  const int koff = (lane >> 4) << 5;   // byte offset 0 or 32
  for (int k0 = 0; k0 < kchunk; k0 += 32) {
    const v16bf a = load_a_frag(A, lda, rowb, kbase + k0, lane);
#pragma unroll
    for (int g = 0; g < 4; ++g) {
      const char* p = buf + (g * 16 + r) * rowStride + k0 * 2 + koff;
      v8bf lo = *(const v8bf*)(p);
      v8bf hi = *(const v8bf*)(p + 16);
      v16bf b;
#pragma unroll
      for (int i = 0; i < 8; ++i) { b[i] = lo[i]; b[i + 8] = hi[i]; }
      acc[g] = __builtin_amdgcn_wmma_f32_16x16x32_bf16(
          false, a, false, b, (short)0, acc[g], false, false);
    }
  }
}

// One GEMM phase: acc += A(B x kdim) @ W(4H x kdim)^T restricted to this
// block's 16 columns of each gate. Weight tile staged through LDS.
DEV_INLINE void gemm_phase(const __bf16* __restrict__ A, int lda, int kdim,
                           const __bf16* __restrict__ W, int ldw,
                           int colb, int rowb, int lane, int wave, int tid,
                           char* smem, v8f acc[4]) {
  const int kchunk    = (kdim < KCHUNK) ? kdim : KCHUNK;
  const int rowStride = kchunk * 2 + 16;
  const unsigned bufBytes = 64u * (unsigned)rowStride;
  const int nc = kdim / kchunk;
  const __bf16* Wt = W + (size_t)colb * ldw;   // tile origin (gate 0)

#if HAVE_TDM
  // NOTE: TDM wants the LDS byte address; addrspace(3) pointers are LDS
  // offsets, recovered here by integer truncation of the shared pointer.
  const unsigned ldsBase = (unsigned)(unsigned long long)smem;
  if (wave == 0) tdm_load_tile(Wt, ldsBase, kchunk, ldw);
  for (int c = 0; c < nc; ++c) {
    if (wave == 0) {
      if (c + 1 < nc) {   // prefetch next chunk, keep it in flight
        tdm_load_tile(Wt + (size_t)(c + 1) * kchunk,
                      ldsBase + (unsigned)((c + 1) & 1) * bufBytes, kchunk, ldw);
        __builtin_amdgcn_s_wait_tensorcnt((short)1);
      } else {
        __builtin_amdgcn_s_wait_tensorcnt((short)0);
      }
    }
    __syncthreads();
    compute_chunk(A, lda, c * kchunk, kchunk, smem + (c & 1) * bufBytes,
                  rowStride, rowb, lane, acc);
    __syncthreads();
  }
#else
  // Fallback: cooperative global->LDS copy with the same padded layout.
  for (int c = 0; c < nc; ++c) {
    char* buf = smem + (c & 1) * bufBytes;
    const int segs = kchunk / 8;               // 16B segments per row
    for (int i = tid; i < 64 * segs; i += 256) {
      const int rowIdx = i / segs, seg = i % segs;
      const int g = rowIdx >> 4, r = rowIdx & 15;
      const __bf16* src = W + ((size_t)g * H_ + colb + r) * ldw
                            + (size_t)c * kchunk + seg * 8;
      *(v8bf*)(buf + rowIdx * rowStride + seg * 16) = *(const v8bf*)src;
    }
    __syncthreads();
    compute_chunk(A, lda, c * kchunk, kchunk, buf, rowStride, rowb, lane, acc);
    __syncthreads();
  }
#endif
}

// ---------------------------------------------------------------------------
// LSTM cell: gates = xin @ Wih^T + hin @ Whh^T + b; update c, write h (bf16).
// Grid: 64 blocks (16 gate-columns), block: 8 waves (16 batch rows each).
// ---------------------------------------------------------------------------
__global__ __launch_bounds__(256)
void lstm_cell_wmma(const __bf16* __restrict__ xin, int ldx, int din,
                    const __bf16* __restrict__ hin,
                    const __bf16* __restrict__ Wih,
                    const __bf16* __restrict__ Whh,
                    const float*  __restrict__ bias,
                    float*        __restrict__ cstate,
                    __bf16*       __restrict__ hout) {
  __shared__ alignas(16) char smem[SMEM_BYTES];
  const int tid  = threadIdx.x;
  const int lane = tid & 31;
  const int wave = tid >> 5;
  const int colb = blockIdx.x << 4;
  const int rowb = wave << 4;

  v8f acc[4] = {};
  gemm_phase(xin, ldx, din, Wih, din, colb, rowb, lane, wave, tid, smem, acc);
  gemm_phase(hin, H_, H_, Whh, H_, colb, rowb, lane, wave, tid, smem, acc);

  // Epilogue. C/D layout: VGPR r, lane L -> row = r + 8*(L>=16), col = L&15.
  const int   n   = colb + (lane & 15);
  const float bi  = bias[n];
  const float bff = bias[H_ + n];
  const float bg  = bias[2 * H_ + n];
  const float bo  = bias[3 * H_ + n];
  const int   mb  = rowb + ((lane >> 4) << 3);
#pragma unroll
  for (int r = 0; r < 8; ++r) {
    const size_t idx = (size_t)(mb + r) * H_ + n;
    const float iv = sigm(acc[0][r] + bi);
    const float fv = sigm(acc[1][r] + bff);
    const float gv = tanhf(acc[2][r] + bg);
    const float ov = sigm(acc[3][r] + bo);
    const float cv = fv * cstate[idx] + iv * gv;
    cstate[idx] = cv;
    hout[idx]   = f2bf(ov * tanhf(cv));
  }
}

// ---------------------------------------------------------------------------
// FC head + teacher forcing: out[:,t,:] = h1 @ fcW^T + fcb;
// dinp = tf[t] ? bf16(y[:,t,:]) : bf16(out). Grid: 4 blocks x 8 waves.
// ---------------------------------------------------------------------------
__global__ __launch_bounds__(256)
void fc_out_wmma(const __bf16* __restrict__ h1,
                 const __bf16* __restrict__ fcW,
                 const float*  __restrict__ fcb,
                 const float*  __restrict__ y,
                 const unsigned char* __restrict__ tfmask,
                 int t,
                 float*        __restrict__ out,
                 __bf16*       __restrict__ dinp) {
  const int lane = threadIdx.x & 31;
  const int wave = threadIdx.x >> 5;
  const int colb = blockIdx.x << 4;
  const int rowb = wave << 4;

  v8f acc = {};
  for (int k0 = 0; k0 < H_; k0 += 32) {
    v16bf a = load_a_frag(h1, H_, rowb, k0, lane);
    // B fragment straight from global: col n K-contiguous, 32B per lane
    const int n    = colb + (lane & 15);
    const int koff = (lane >> 4) << 4;
    const __bf16* p = fcW + (size_t)n * H_ + k0 + koff;
    v8bf lo = *(const v8bf*)(p);
    v8bf hi = *(const v8bf*)(p + 8);
    v16bf b;
#pragma unroll
    for (int i = 0; i < 8; ++i) { b[i] = lo[i]; b[i + 8] = hi[i]; }
    acc = __builtin_amdgcn_wmma_f32_16x16x32_bf16(
        false, a, false, b, (short)0, acc, false, false);
  }

  const int   n     = colb + (lane & 15);
  const float bn    = fcb[n];
  const int   force = tfmask[t] != 0;
  const int   mb    = rowb + ((lane >> 4) << 3);
#pragma unroll
  for (int r = 0; r < 8; ++r) {
    const int    m    = mb + r;
    const size_t oidx = ((size_t)m * T_ + t) * D_ + n;  // (B,T,D) row-major
    const float  v    = acc[r] + bn;
    out[oidx] = v;
    const float dv = force ? y[oidx] : v;
    dinp[(size_t)m * D_ + n] = f2bf(dv);
  }
}

__global__ void cvt_f32_bf16(const float* __restrict__ s,
                             __bf16* __restrict__ d, int n) {
  int i = blockIdx.x * blockDim.x + threadIdx.x;
  if (i < n) d[i] = f2bf(s[i]);
}

// ---------------------------------------------------------------------------
extern "C" void kernel_launch(void* const* d_in, const int* in_sizes, int n_in,
                              void* d_out, int out_size, void* d_ws,
                              size_t ws_size, hipStream_t stream) {
  (void)in_sizes; (void)n_in; (void)out_size; (void)ws_size;

  const float* x     = (const float*)d_in[0];
  const float* y     = (const float*)d_in[1];
  const unsigned char* tf = (const unsigned char*)d_in[2];  // jax bool mask
  const float* eWih0 = (const float*)d_in[3];
  const float* eWhh0 = (const float*)d_in[4];
  const float* eb0   = (const float*)d_in[5];
  const float* eWih1 = (const float*)d_in[6];
  const float* eWhh1 = (const float*)d_in[7];
  const float* eb1   = (const float*)d_in[8];
  const float* dWih0 = (const float*)d_in[9];
  const float* dWhh0 = (const float*)d_in[10];
  const float* db0   = (const float*)d_in[11];
  const float* dWih1 = (const float*)d_in[12];
  const float* dWhh1 = (const float*)d_in[13];
  const float* db1   = (const float*)d_in[14];
  const float* fcW   = (const float*)d_in[15];
  const float* fcb   = (const float*)d_in[16];
  float* out = (float*)d_out;

  // ---- workspace carve (~58 MB) ----
  char* wp = (char*)d_ws;
  auto carve = [&](size_t bytes) -> char* {
    char* r = wp;
    wp += (bytes + 255) & ~(size_t)255;
    return r;
  };
  __bf16* xbf    = (__bf16*)carve((size_t)B_ * T_ * D_ * 2);
  __bf16* w_e0ih = (__bf16*)carve((size_t)4 * H_ * D_ * 2);
  __bf16* w_e0hh = (__bf16*)carve((size_t)4 * H_ * H_ * 2);
  __bf16* w_e1ih = (__bf16*)carve((size_t)4 * H_ * H_ * 2);
  __bf16* w_e1hh = (__bf16*)carve((size_t)4 * H_ * H_ * 2);
  __bf16* w_d0ih = (__bf16*)carve((size_t)4 * H_ * D_ * 2);
  __bf16* w_d0hh = (__bf16*)carve((size_t)4 * H_ * H_ * 2);
  __bf16* w_d1ih = (__bf16*)carve((size_t)4 * H_ * H_ * 2);
  __bf16* w_d1hh = (__bf16*)carve((size_t)4 * H_ * H_ * 2);
  __bf16* w_fc   = (__bf16*)carve((size_t)D_ * H_ * 2);
  __bf16* h0b[2] = { (__bf16*)carve((size_t)B_ * H_ * 2),
                     (__bf16*)carve((size_t)B_ * H_ * 2) };
  __bf16* h1b[2] = { (__bf16*)carve((size_t)B_ * H_ * 2),
                     (__bf16*)carve((size_t)B_ * H_ * 2) };
  float* c0   = (float*)carve((size_t)B_ * H_ * 4);
  float* c1   = (float*)carve((size_t)B_ * H_ * 4);
  __bf16* dinp = (__bf16*)carve((size_t)B_ * D_ * 2);

  auto cvt = [&](const float* s, __bf16* d, size_t n) {
    cvt_f32_bf16<<<(unsigned)((n + 255) / 256), 256, 0, stream>>>(s, d, (int)n);
  };
  cvt(x, xbf, (size_t)B_ * T_ * D_);
  cvt(eWih0, w_e0ih, (size_t)4 * H_ * D_);
  cvt(eWhh0, w_e0hh, (size_t)4 * H_ * H_);
  cvt(eWih1, w_e1ih, (size_t)4 * H_ * H_);
  cvt(eWhh1, w_e1hh, (size_t)4 * H_ * H_);
  cvt(dWih0, w_d0ih, (size_t)4 * H_ * D_);
  cvt(dWhh0, w_d0hh, (size_t)4 * H_ * H_);
  cvt(dWih1, w_d1ih, (size_t)4 * H_ * H_);
  cvt(dWhh1, w_d1hh, (size_t)4 * H_ * H_);
  cvt(fcW, w_fc, (size_t)D_ * H_);

  hipMemsetAsync(h0b[0], 0, (size_t)B_ * H_ * 2, stream);
  hipMemsetAsync(h1b[0], 0, (size_t)B_ * H_ * 2, stream);
  hipMemsetAsync(c0, 0, (size_t)B_ * H_ * 4, stream);
  hipMemsetAsync(c1, 0, (size_t)B_ * H_ * 4, stream);

  const dim3 cgrid(H_ / 16), cblk(256);

  // ---- encoder: interleave layer0/layer1 per timestep ----
  for (int t = 0; t < T_; ++t) {
    const int p = t & 1;
    lstm_cell_wmma<<<cgrid, cblk, 0, stream>>>(
        xbf + (size_t)t * D_, T_ * D_, D_, h0b[p],
        w_e0ih, w_e0hh, eb0, c0, h0b[p ^ 1]);
    lstm_cell_wmma<<<cgrid, cblk, 0, stream>>>(
        h0b[p ^ 1], H_, H_, h1b[p],
        w_e1ih, w_e1hh, eb1, c1, h1b[p ^ 1]);
  }

  // ---- decoder (states carry over; T even keeps parity at t&1) ----
  for (int t = 0; t < T_; ++t) {
    const int p = t & 1;
    const __bf16* xin = (t == 0) ? (xbf + (size_t)(T_ - 1) * D_) : dinp;
    const int ldx     = (t == 0) ? (T_ * D_) : D_;
    lstm_cell_wmma<<<cgrid, cblk, 0, stream>>>(
        xin, ldx, D_, h0b[p], w_d0ih, w_d0hh, db0, c0, h0b[p ^ 1]);
    lstm_cell_wmma<<<cgrid, cblk, 0, stream>>>(
        h0b[p ^ 1], H_, H_, h1b[p], w_d1ih, w_d1hh, db1, c1, h1b[p ^ 1]);
    fc_out_wmma<<<dim3(D_ / 16), cblk, 0, stream>>>(
        h1b[p ^ 1], w_fc, fcb, y, tf, t, out, dinp);
  }
}
